// TriangleSelfAttention_73650099192410
// MI455X (gfx1250) — matrix operand order repaired and checked by reading, not compiled, not run on hardware
//
#include <hip/hip_runtime.h>
#include <hip/hip_bf16.h>

typedef float    v8f  __attribute__((ext_vector_type(8)));
typedef float    v2f  __attribute__((ext_vector_type(2)));
typedef _Float16 v16h __attribute__((ext_vector_type(16)));
typedef _Float16 v8h  __attribute__((ext_vector_type(8)));

#if defined(__has_builtin)
#if __has_builtin(__builtin_amdgcn_wmma_f32_16x16x4_f32)
#define HAS_WMMA_F32X4 1
#endif
#endif
#ifndef HAS_WMMA_F32X4
#define HAS_WMMA_F32X4 0
#endif

#define NTOK 256
#define DZv  64
#define EPS_MASK 50000.0f

// ---- LDS layout (bytes) --------------------------------------------------
#define OFF_ZLN   0                               // f16 [256][64]  (aliased by gprod later)
#define OFF_WT    (OFF_ZLN  + 256*64*2)           // f16 [128][64]  W{q,k,v,g} transposed [col][k]
#define OFF_WOT   (OFF_WT   + 128*64*2)           // f16 [64][32]   Wo transposed [outc][k]
#define OFF_QS    (OFF_WOT  + 64*32*2)            // f32 [256][32]  q * 1/sqrt(DC)
#define OFF_KS    (OFF_QS   + 256*32*4)           // f32 [256][32]
#define OFF_VT    (OFF_KS   + 256*32*4)           // f16 [48][256]  v transposed [c][j], rows 32..47 zero pad
#define OFF_GATE  (OFF_VT   + 48*256*2)           // f32 [256][32]  sigmoid(gate)
#define OFF_VAVG  (OFF_GATE + 256*32*4)           // f32 [256][32]
#define OFF_S     (OFF_VAVG + 256*32*4)           // f16 [8 waves][16][256] softmax weights
#define OFF_JB    (OFF_S    + 8*16*256*2)         // f32 [256] column mask bias
#define OFF_BALL  (OFF_JB   + 256*4)              // f32 [128] b{q,k,v,g}
#define OFF_BOS   (OFF_BALL + 128*4)              // f32 [64]  bo
#define SMEM_BYTES (OFF_BOS + 64*4)               // = 276224 bytes  (<= 320 KB/WGP)

__device__ __forceinline__ v16h ld_a16(const _Float16* p0, const _Float16* p1) {
  union { uint4 q[2]; v16h h; } u;
  u.q[0] = *(const uint4*)p0;
  u.q[1] = *(const uint4*)p1;
  return u.h;
}

__device__ __forceinline__ float fast_rcp(float x) { return __builtin_amdgcn_rcpf(x); }

__global__ __launch_bounds__(256, 1)
void tsa_fused(const float* __restrict__ z,    const float* __restrict__ mask,
               const float* __restrict__ msa,  const float* __restrict__ ln_g,
               const float* __restrict__ ln_b,
               const float* __restrict__ Wq, const float* __restrict__ bq,
               const float* __restrict__ Wk, const float* __restrict__ bk,
               const float* __restrict__ Wv, const float* __restrict__ bv,
               const float* __restrict__ Wg, const float* __restrict__ bg,
               const float* __restrict__ Wo, const float* __restrict__ bo,
               float* __restrict__ out)
{
  extern __shared__ char smem[];
  _Float16* zln  = (_Float16*)(smem + OFF_ZLN);
  _Float16* wt   = (_Float16*)(smem + OFF_WT);
  _Float16* wot  = (_Float16*)(smem + OFF_WOT);
  float*    qs   = (float*)   (smem + OFF_QS);
  float*    ksm  = (float*)   (smem + OFF_KS);
  _Float16* vt   = (_Float16*)(smem + OFF_VT);
  float*    gate = (float*)   (smem + OFF_GATE);
  float*    vavg = (float*)   (smem + OFF_VAVG);
  _Float16* Sbuf = (_Float16*)(smem + OFF_S);
  float*    jb   = (float*)   (smem + OFF_JB);
  float*    ball = (float*)   (smem + OFF_BALL);
  float*    bos  = (float*)   (smem + OFF_BOS);

  const int tid = threadIdx.x;
  const int n   = blockIdx.x;          // attention row
  const int wv  = tid >> 5;            // wave 0..7
  const int ln  = tid & 31;            // lane
  const int lh  = ln >> 4;             // lane half
  const int ll  = ln & 15;             // lane low
  const int wvu = __builtin_amdgcn_readfirstlane(wv);   // wave id as SGPR (uniform branches)

  // ---------------- Phase 0: stage weights, biases, mask bias -------------
  for (int idx = tid; idx < 128 * 64; idx += 256) {
    int k = idx >> 7;                  // 0..63 (consecutive tids -> consecutive cols: coalesced)
    int c = idx & 127;                 // 0..127
    const float* W = (c < 32) ? Wq : (c < 64) ? Wk : (c < 96) ? Wv : Wg;
    wt[c * 64 + k] = (_Float16)W[k * 32 + (c & 31)];
  }
  for (int idx = tid; idx < 64 * 32; idx += 256) {
    int k = idx >> 6; int oc = idx & 63;
    wot[oc * 32 + k] = (_Float16)Wo[k * 64 + oc];
  }
  if (tid < 128)
    ball[tid] = (tid < 32) ? bq[tid] : (tid < 64) ? bk[tid - 32]
              : (tid < 96) ? bv[tid - 64] : bg[tid - 96];
  if (tid < 64) bos[tid] = bo[tid];
  {
    float mj = mask[tid];
    float ms = msa[(size_t)n * NTOK + tid];
    jb[tid] = -(1.0f - mj) * EPS_MASK - (1.0f - ms) * EPS_MASK;
  }
  for (int idx = tid; idx < 16 * 256; idx += 256)   // zero V padding rows 32..47
    vt[32 * 256 + idx] = (_Float16)0.0f;

  // ---------------- Phase 1: LayerNorm (one token per thread) -------------
  {
    const float* zp = z + ((size_t)n * NTOK + tid) * DZv;
    float buf[DZv];
    float s = 0.0f;
    #pragma unroll
    for (int t = 0; t < DZv / 4; ++t) {
      float4 f = ((const float4*)zp)[t];
      buf[4*t+0] = f.x; buf[4*t+1] = f.y; buf[4*t+2] = f.z; buf[4*t+3] = f.w;
      s += f.x + f.y + f.z + f.w;
    }
    float mu = s * (1.0f / DZv);
    float vs = 0.0f;
    #pragma unroll
    for (int t = 0; t < DZv; ++t) { float d = buf[t] - mu; vs += d * d; }
    float rstd = rsqrtf(vs * (1.0f / DZv) + 1e-5f);
    #pragma unroll
    for (int t = 0; t < DZv; ++t)
      zln[tid * DZv + t] = (_Float16)(((buf[t] - mu) * rstd) * ln_g[t] + ln_b[t]);
  }
  __syncthreads();

  // ---------------- Phase 2: q/k/v/gate projections (WMMA f16, K=64) ------
  {
    const int colg = wvu * 16 + ll;    // wave -> 16-column tile of the 128 combined cols
    v16h bfr[2];
    #pragma unroll
    for (int k2 = 0; k2 < 2; ++k2)     // B: col per lane, sequential 16 K per half
      bfr[k2] = *(const v16h*)&wt[colg * 64 + k2 * 32 + lh * 16];
    const float bias = ball[colg];
    const float scal = 0.35355339059327373f;   // 1/sqrt(DC)

    for (int it = 0; it < 16; ++it) {
      const int i0 = it * 16;
      v8f acc = {};
      #pragma unroll
      for (int k2 = 0; k2 < 2; ++k2) {
        const int kb = k2 * 32 + lh * 16;      // A: sequential 16 K per half
        v16h a = ld_a16(&zln[(i0 + ll) * 64 + kb], &zln[(i0 + ll) * 64 + kb + 8]);
        acc = __builtin_amdgcn_wmma_f32_16x16x32_f16(false, a, false, bfr[k2],
                                                     (short)0, acc, false, false);
      }
      // D: col = colg (lane), row = i0 + lh*8 + v.  Scalar branch on wave id.
      if (wvu < 2) {                       // cols 0..31: q
        #pragma unroll
        for (int v = 0; v < 8; ++v)
          qs[(i0 + lh * 8 + v) * 32 + colg] = (acc[v] + bias) * scal;
      } else if (wvu < 4) {                // cols 32..63: k
        #pragma unroll
        for (int v = 0; v < 8; ++v)
          ksm[(i0 + lh * 8 + v) * 32 + (colg - 32)] = acc[v] + bias;
      } else if (wvu < 6) {                // cols 64..95: v (store transposed f16)
        v8h hv;
        #pragma unroll
        for (int v = 0; v < 8; ++v) hv[v] = (_Float16)(acc[v] + bias);
        *(v8h*)&vt[(colg - 64) * 256 + i0 + lh * 8] = hv;
      } else {                             // cols 96..127: sigmoid gate
        #pragma unroll
        for (int v = 0; v < 8; ++v) {
          float e = __expf(-(acc[v] + bias));
          gate[(i0 + lh * 8 + v) * 32 + (colg - 96)] = fast_rcp(1.0f + e);
        }
      }
    }
  }
  __syncthreads();

  // ---------------- Phase 3: attention, 8 (i-tile, head) units per wave ---
  _Float16* Sw = Sbuf + wvu * 16 * 256;
  for (int u8 = 0; u8 < 8; ++u8) {
    const int u   = wvu * 8 + u8;
    const int it  = u >> 2, h = u & 3;
    const int i0  = it * 16, c0h = h * 8;

    v8f fac[16];
#if HAS_WMMA_F32X4
    v2f qfr[2];
    #pragma unroll
    for (int kk = 0; kk < 2; ++kk)     // A f32 16x4: row=i0+ll, K = c0h + 4kk + 2*lh + {0,1}
      qfr[kk] = *(const v2f*)&qs[(i0 + ll) * 32 + c0h + kk * 4 + lh * 2];
    #pragma unroll
    for (int jt = 0; jt < 16; ++jt) {
      const int j0 = jt * 16;
      v8f acc = {};
      #pragma unroll
      for (int kk = 0; kk < 2; ++kk) {
        v2f bfr = *(const v2f*)&ksm[(j0 + ll) * 32 + c0h + kk * 4 + lh * 2];
        acc = __builtin_amdgcn_wmma_f32_16x16x4_f32(false, qfr[kk], false, bfr,
                                                    (short)0, acc, false, false);
      }
      const float b = jb[j0 + ll];
      #pragma unroll
      for (int v = 0; v < 8; ++v) acc[v] += b;
      fac[jt] = acc;
    }
#else
    float qreg[8][8];
    #pragma unroll
    for (int v = 0; v < 8; ++v)
      #pragma unroll
      for (int c = 0; c < 8; ++c)
        qreg[v][c] = qs[(i0 + lh * 8 + v) * 32 + c0h + c];
    #pragma unroll
    for (int jt = 0; jt < 16; ++jt) {
      float kv[8];
      #pragma unroll
      for (int c = 0; c < 8; ++c) kv[c] = ksm[(jt * 16 + ll) * 32 + c0h + c];
      const float b = jb[jt * 16 + ll];
      #pragma unroll
      for (int v = 0; v < 8; ++v) {
        float s = b;
        #pragma unroll
        for (int c = 0; c < 8; ++c) s += qreg[v][c] * kv[c];
        fac[jt][v] = s;
      }
    }
#endif
    // softmax over j: row = (lh, v) fixed per lane; columns spread over ll and jt
    float mx[8], sm[8], inv[8];
    #pragma unroll
    for (int v = 0; v < 8; ++v) mx[v] = fac[0][v];
    #pragma unroll
    for (int jt = 1; jt < 16; ++jt)
      #pragma unroll
      for (int v = 0; v < 8; ++v) mx[v] = fmaxf(mx[v], fac[jt][v]);
    #pragma unroll
    for (int off = 1; off < 16; off <<= 1)
      #pragma unroll
      for (int v = 0; v < 8; ++v) mx[v] = fmaxf(mx[v], __shfl_xor(mx[v], off, 32));
    #pragma unroll
    for (int v = 0; v < 8; ++v) sm[v] = 0.0f;
    #pragma unroll
    for (int jt = 0; jt < 16; ++jt)
      #pragma unroll
      for (int v = 0; v < 8; ++v) {
        float e = __expf(fac[jt][v] - mx[v]);
        fac[jt][v] = e; sm[v] += e;
      }
    #pragma unroll
    for (int off = 1; off < 16; off <<= 1)
      #pragma unroll
      for (int v = 0; v < 8; ++v) sm[v] += __shfl_xor(sm[v], off, 32);
    #pragma unroll
    for (int v = 0; v < 8; ++v) inv[v] = fast_rcp(sm[v]);

    #pragma unroll
    for (int jt = 0; jt < 16; ++jt)
      #pragma unroll
      for (int v = 0; v < 8; ++v)
        Sw[(lh * 8 + v) * 256 + jt * 16 + ll] = (_Float16)(fac[jt][v] * inv[v]);

    // AV: (16 x 256) x (256 x 16[c, 8 valid]) via f16 WMMA, K=32 per step
    v8f vac = {};
    #pragma unroll
    for (int kt = 0; kt < 8; ++kt) {
      const int j0 = kt * 32;
      v16h a  = ld_a16(&Sw[ll * 256 + j0 + lh * 16], &Sw[ll * 256 + j0 + lh * 16 + 8]);
      v16h bb = *(const v16h*)&vt[(c0h + ll) * 256 + j0 + lh * 16];
      vac = __builtin_amdgcn_wmma_f32_16x16x32_f16(false, a, false, bb,
                                                   (short)0, vac, false, false);
    }
    if (ll < 8) {
      #pragma unroll
      for (int v = 0; v < 8; ++v)
        vavg[(i0 + lh * 8 + v) * 32 + c0h + ll] = vac[v];
    }
  }
  __syncthreads();

  // ---------------- Phase 4: gate multiply, output projection --------------
  _Float16* gp = zln;                    // alias: zln no longer needed
  for (int idx = tid; idx < 256 * 32; idx += 256)
    gp[idx] = (_Float16)(vavg[idx] * gate[idx]);
  __syncthreads();

  for (int t8 = 0; t8 < 8; ++t8) {
    const int t  = wvu * 8 + t8;
    const int it = t >> 2, ct = t & 3;
    const int i0 = it * 16;
    const int oc = ct * 16 + ll;
    v16h bb = *(const v16h*)&wot[oc * 32 + lh * 16];
    v16h a  = ld_a16(&gp[(i0 + ll) * 32 + lh * 16], &gp[(i0 + ll) * 32 + lh * 16 + 8]);
    v8f acc = {};
    acc = __builtin_amdgcn_wmma_f32_16x16x32_f16(false, a, false, bb,
                                                 (short)0, acc, false, false);
    const float bias = bos[oc];
    #pragma unroll
    for (int v = 0; v < 8; ++v)
      out[((size_t)n * NTOK + i0 + lh * 8 + v) * DZv + oc] = acc[v] + bias;
  }
}

extern "C" void kernel_launch(void* const* d_in, const int* in_sizes, int n_in,
                              void* d_out, int out_size, void* d_ws, size_t ws_size,
                              hipStream_t stream) {
  (void)in_sizes; (void)n_in; (void)out_size; (void)d_ws; (void)ws_size;
  const float* z    = (const float*)d_in[0];
  const float* mask = (const float*)d_in[1];
  const float* msa  = (const float*)d_in[2];
  const float* ln_g = (const float*)d_in[3];
  const float* ln_b = (const float*)d_in[4];
  const float* Wq   = (const float*)d_in[5];
  const float* bq   = (const float*)d_in[6];
  const float* Wk   = (const float*)d_in[7];
  const float* bk   = (const float*)d_in[8];
  const float* Wv   = (const float*)d_in[9];
  const float* bv   = (const float*)d_in[10];
  const float* Wg   = (const float*)d_in[11];
  const float* bg   = (const float*)d_in[12];
  const float* Wo   = (const float*)d_in[13];
  const float* bo   = (const float*)d_in[14];
  float* outp = (float*)d_out;

  (void)hipFuncSetAttribute((const void*)tsa_fused,
                            hipFuncAttributeMaxDynamicSharedMemorySize, SMEM_BYTES);
  tsa_fused<<<dim3(NTOK), dim3(256), SMEM_BYTES, stream>>>(
      z, mask, msa, ln_g, ln_b, Wq, bq, Wk, bk, Wv, bv, Wg, bg, Wo, bo, outp);
}
